// man_LSTM_11940009083295
// MI455X (gfx1250) — matrix operand order repaired and checked by reading, not compiled, not run on hardware
//
#include <hip/hip_runtime.h>
#include <hip/hip_bf16.h>
#include <stddef.h>

typedef __attribute__((ext_vector_type(16))) __bf16 v16bf;
typedef __attribute__((ext_vector_type(8)))  __bf16 v8bf;
typedef __attribute__((ext_vector_type(8)))  float  v8f;
typedef int v4i __attribute__((vector_size(16)));

#define AS1 __attribute__((address_space(1)))
#define AS3 __attribute__((address_space(3)))

#define B_    64
#define T_    512
#define I_    512
#define H_    512
#define NTOT  2048                // 4 * H_
#define NT_   128                 // NTOT / 16   (N tiles)
#define FRAG_ELEMS 512            // 32 lanes * 16 bf16 per fragment (1024 B)

#define WMMA_BF16 __builtin_amdgcn_wmma_f32_16x16x32_bf16

__device__ __forceinline__ float sigm_(float x)  { return 1.0f / (1.0f + __expf(-x)); }
__device__ __forceinline__ float tanh_f(float x) { return 2.0f / (1.0f + __expf(-2.0f * x)) - 1.0f; }

// ---------------------------------------------------------------------------
// One-time: swizzle combined B = [W^T ; U^T] (K=1024 x N=2048) into WMMA
// B-fragment layout. Fragment (kt, nt) holds a 32(K) x 16(N) bf16 tile,
// stored lane-major: lane L owns 16 contiguous bf16 (32 B).
//   lane L: n = nt*16 + (L&15), element e<8 holds K = kt*32 + (L>>4)*8 + e, e>=8 holds K = kt*32 + 16 + (L>>4)*8 + (e-8).
// Gate order along N matches the reference concat: [i, f, o, c].
// ---------------------------------------------------------------------------
__global__ void swizzle_B_kernel(const float* __restrict__ Wi, const float* __restrict__ Ui,
                                 const float* __restrict__ Wf, const float* __restrict__ Uf,
                                 const float* __restrict__ Wo, const float* __restrict__ Uo,
                                 const float* __restrict__ Wc, const float* __restrict__ Uc,
                                 __bf16* __restrict__ Bsw) {
    int tid  = blockIdx.x * blockDim.x + threadIdx.x;   // 32*128*32 = 131072 threads
    int lane = tid & 31;
    int frag = tid >> 5;                                // 4096 fragments
    int nt   = frag & (NT_ - 1);
    int kt   = frag >> 7;                               // 0..31
    int half = lane >> 4;
    int n    = nt * 16 + (lane & 15);                   // 0..2047
    int g    = n >> 9;                                  // gate slot 0..3 = i,f,o,c
    int jn   = n & (H_ - 1);
    const float* Wg = (g == 0) ? Wi : (g == 1) ? Wf : (g == 2) ? Wo : Wc;
    const float* Ug = (g == 0) ? Ui : (g == 1) ? Uf : (g == 2) ? Uo : Uc;
    __bf16* dst = Bsw + (size_t)frag * FRAG_ELEMS + lane * 16;
    int kbase = kt * 32 + half * 8;                     // within 32-aligned tile -> never mixes W/U
    const float* src = (kbase < I_) ? (Wg + (size_t)jn * I_ + kbase)
                                    : (Ug + (size_t)jn * H_ + (kbase - I_));
#pragma unroll
    for (int e = 0; e < 16; ++e) dst[e] = (__bf16)src[e < 8 ? e : e + 8];
}

// ---------------------------------------------------------------------------
// One-time: swizzle x [B, T, I] f32 into WMMA A-fragment layout (bf16).
// Fragment id = ((t*16 + kt)*4 + mtile). Lane L owns 16 contiguous bf16:
//   m = mtile*16 + (L&15), half = L>>4
//   elements 0..7  -> K = kt*32 + half*8 + e
//   elements 8..15 -> K = kt*32 + 16 + half*8 + (e-8)
// ---------------------------------------------------------------------------
__global__ void swizzle_X_kernel(const float* __restrict__ x, __bf16* __restrict__ xf) {
    int tid  = blockIdx.x * blockDim.x + threadIdx.x;   // 512*16*4*32 = 1,048,576
    int lane = tid & 31;
    int frag = tid >> 5;                                // 32768 fragments
    int mtile = frag & 3;
    int kt    = (frag >> 2) & 15;
    int t     = frag >> 6;
    int half  = lane >> 4;
    int m     = mtile * 16 + (lane & 15);
    const float* src = x + ((size_t)m * T_ + t) * I_ + kt * 32 + half * 8;
    __bf16* dst = xf + (size_t)frag * FRAG_ELEMS + lane * 16;
#pragma unroll
    for (int e = 0; e < 8; ++e) dst[e]     = (__bf16)src[e];
#pragma unroll
    for (int e = 0; e < 8; ++e) dst[8 + e] = (__bf16)src[16 + e];
}

// ---------------------------------------------------------------------------
// One-time: init h (bf16 row-major [64,512]), c (f32), and the barrier counter.
// ---------------------------------------------------------------------------
__global__ void init_state_kernel(const float* __restrict__ h0, const float* __restrict__ c0,
                                  __bf16* __restrict__ hbf, float* __restrict__ cbuf,
                                  unsigned* __restrict__ cnt) {
    int i = blockIdx.x * blockDim.x + threadIdx.x;
    if (i == 0) *cnt = 0u;
    if (i < B_ * H_) { hbf[i] = (__bf16)h0[i]; cbuf[i] = c0[i]; }
}

// ---------------------------------------------------------------------------
// Persistent LSTM: 32 WGs x 128 threads = 128 waves.
// All 4 waves of a WG share j = blockIdx.x (hidden 16-col tile); mtile = waveId.
// The WG's 128 KB B slice (32 kt x 4 gates x 1 KB fragments) is staged into
// LDS once via async global->LDS copies, then every step reads B via
// ds_load_b128 while x/h stream through the vmem pipe.
// ---------------------------------------------------------------------------
__global__ __launch_bounds__(128)
void lstm_persistent_kernel(const __bf16* __restrict__ xf,
                            const __bf16* __restrict__ Bsw,
                            const float* __restrict__ bi_p, const float* __restrict__ bf_p,
                            const float* __restrict__ bo_p, const float* __restrict__ bc_p,
                            __bf16* __restrict__ h0buf, __bf16* __restrict__ h1buf,
                            float* __restrict__ cbuf,
                            float* __restrict__ out, float* __restrict__ outH,
                            float* __restrict__ outC,
                            unsigned* __restrict__ cnt) {
    extern __shared__ __bf16 Blds[];          // 131072 B: fragment (kt,g) at (kt*4+g)*512

    const int lane  = threadIdx.x & 31;
    const int mtile = threadIdx.x >> 5;       // 0..3 (wave id in WG)
    const int j     = blockIdx.x;             // 0..31 hidden tile
    const int half  = lane >> 4;
    const int ml    = lane & 15;
    const int n     = j * 16 + ml;            // hidden column this lane owns in C/D
    const int mrow  = mtile * 16 + ml;        // batch row this lane loads for A(h)

    // ---- Stage this WG's B slice into LDS with async global->LDS copies ----
    {
        const int i  = threadIdx.x;           // thread == one (kt, gate) fragment
        const int kt = i >> 2;
        const int g  = i & 3;
        const __bf16* gsrc = Bsw + ((size_t)kt * NT_ + (size_t)g * 32 + (size_t)j) * FRAG_ELEMS;
        __bf16* ldst = Blds + (size_t)(kt * 4 + g) * FRAG_ELEMS;
#pragma unroll
        for (int q = 0; q < 64; ++q) {        // 64 x b128 = 1024 B per lane
            __builtin_amdgcn_global_load_async_to_lds_b128(
                (AS1 v4i*)(gsrc + q * 8),
                (AS3 v4i*)(ldst + q * 8), 0, 0);
        }
        __builtin_amdgcn_s_wait_asynccnt(0);
        __syncthreads();
    }

    const float bias0 = bi_p[n];
    const float bias1 = bf_p[n];
    const float bias2 = bo_p[n];
    const float bias3 = bc_p[n];

    const __bf16* blane = Blds + (size_t)lane * 16;   // per-lane base within fragments

    for (int t = 0; t < T_; ++t) {
        const __bf16* hread  = (t & 1) ? h1buf : h0buf;
        __bf16*       hwrite = (t & 1) ? h0buf : h1buf;

        // Prefetch next step's x fragments into cache while this step computes.
        if (t + 1 < T_) {
            const char* nxt = (const char*)(xf + (((size_t)(t + 1) * 16) * 4 + mtile) * FRAG_ELEMS)
                              + (size_t)lane * 32;
#pragma unroll
            for (int kt = 0; kt < 16; ++kt)
                __builtin_prefetch(nxt + (size_t)kt * 4096, 0, 3);
        }

        v8f acc0, acc1, acc2, acc3;
#pragma unroll
        for (int r = 0; r < 8; ++r) { acc0[r] = bias0; acc1[r] = bias1; acc2[r] = bias2; acc3[r] = bias3; }

        // ---- K-steps 0..15: x contribution (A from global, B from LDS) ----
        const __bf16* xfr = xf + (((size_t)t * 16) * 4 + mtile) * FRAG_ELEMS + (size_t)lane * 16;
#pragma unroll 4
        for (int kt = 0; kt < 16; ++kt) {
            v16bf a = *(const v16bf*)xfr;
            xfr += 4 * FRAG_ELEMS;
            const __bf16* bp = blane + (size_t)(kt * 4) * FRAG_ELEMS;
            v16bf b0 = *(const v16bf*)(bp);
            v16bf b1 = *(const v16bf*)(bp + FRAG_ELEMS);
            v16bf b2 = *(const v16bf*)(bp + 2 * FRAG_ELEMS);
            v16bf b3 = *(const v16bf*)(bp + 3 * FRAG_ELEMS);
            acc0 = WMMA_BF16(false, a, false, b0, (short)0, acc0, false, false);
            acc1 = WMMA_BF16(false, a, false, b1, (short)0, acc1, false, false);
            acc2 = WMMA_BF16(false, a, false, b2, (short)0, acc2, false, false);
            acc3 = WMMA_BF16(false, a, false, b3, (short)0, acc3, false, false);
        }

        // ---- K-steps 16..31: recurrent h contribution ----
        const __bf16* hrow = hread + (size_t)mrow * H_;
#pragma unroll 4
        for (int kt2 = 0; kt2 < 16; ++kt2) {
            int kb = kt2 * 32 + half * 8;
            v8bf alo = *(const v8bf*)(hrow + kb);
            v8bf ahi = *(const v8bf*)(hrow + kb + 16);
            v16bf a = __builtin_shufflevector(alo, ahi, 0,1,2,3,4,5,6,7,8,9,10,11,12,13,14,15);
            const __bf16* bp = blane + (size_t)((16 + kt2) * 4) * FRAG_ELEMS;
            v16bf b0 = *(const v16bf*)(bp);
            v16bf b1 = *(const v16bf*)(bp + FRAG_ELEMS);
            v16bf b2 = *(const v16bf*)(bp + 2 * FRAG_ELEMS);
            v16bf b3 = *(const v16bf*)(bp + 3 * FRAG_ELEMS);
            acc0 = WMMA_BF16(false, a, false, b0, (short)0, acc0, false, false);
            acc1 = WMMA_BF16(false, a, false, b1, (short)0, acc1, false, false);
            acc2 = WMMA_BF16(false, a, false, b2, (short)0, acc2, false, false);
            acc3 = WMMA_BF16(false, a, false, b3, (short)0, acc3, false, false);
        }

        // ---- Elementwise LSTM cell epilogue (C/D layout: row = half*8 + r) ----
#pragma unroll
        for (int r = 0; r < 8; ++r) {
            int   mg   = mtile * 16 + half * 8 + r;
            float gi   = sigm_(acc0[r]);
            float gf   = sigm_(acc1[r]);
            float go   = sigm_(acc2[r]);
            float gg   = tanh_f(acc3[r]);
            int   ci   = mg * H_ + n;                 // uniquely owned by this lane
            float cold = cbuf[ci];
            float cnew = gf * cold + gi * gg;
            float hnew = go * tanh_f(cnew);
            cbuf[ci]   = cnew;
            hwrite[ci] = (__bf16)hnew;
            out[((size_t)mg * T_ + t) * H_ + n] = hnew;
            if (t == T_ - 1) { outH[ci] = hnew; outC[ci] = cnew; }
        }

        // ---- Grid-wide barrier: all h/c writes visible before next step ----
        __threadfence();
        __syncthreads();
        if (threadIdx.x == 0) {
            __hip_atomic_fetch_add(cnt, 1u, __ATOMIC_RELEASE, __HIP_MEMORY_SCOPE_AGENT);
            unsigned target = (unsigned)(t + 1) * gridDim.x;
            while (__hip_atomic_load(cnt, __ATOMIC_ACQUIRE, __HIP_MEMORY_SCOPE_AGENT) < target) {
                __builtin_amdgcn_s_sleep(2);
            }
        }
        __syncthreads();
    }
}

// ---------------------------------------------------------------------------
// setup_inputs order:
// 0:x 1:h0 2:c0 3:W_i 4:U_i 5:b_i 6:W_f 7:U_f 8:b_f 9:W_c 10:U_c 11:b_c
// 12:W_o 13:U_o 14:b_o
// d_out = outputs[64,512,512] f32 ++ h_T[64,512] ++ c_T[64,512]
// ---------------------------------------------------------------------------
extern "C" void kernel_launch(void* const* d_in, const int* in_sizes, int n_in,
                              void* d_out, int out_size, void* d_ws, size_t ws_size,
                              hipStream_t stream) {
    const float* x  = (const float*)d_in[0];
    const float* h0 = (const float*)d_in[1];
    const float* c0 = (const float*)d_in[2];
    const float* Wi = (const float*)d_in[3];
    const float* Ui = (const float*)d_in[4];
    const float* bi = (const float*)d_in[5];
    const float* Wf = (const float*)d_in[6];
    const float* Uf = (const float*)d_in[7];
    const float* bf = (const float*)d_in[8];
    const float* Wc = (const float*)d_in[9];
    const float* Uc = (const float*)d_in[10];
    const float* bc = (const float*)d_in[11];
    const float* Wo = (const float*)d_in[12];
    const float* Uo = (const float*)d_in[13];
    const float* bo = (const float*)d_in[14];

    char* ws = (char*)d_ws;
    unsigned* cnt = (unsigned*)ws;                                  // 256 B slot
    __bf16*   Bsw = (__bf16*)(ws + 256);                            // 4 MB
    __bf16*   xf  = (__bf16*)(ws + 256 + ((size_t)4 << 20));        // 32 MB
    __bf16*   h0b = (__bf16*)(ws + 256 + ((size_t)36 << 20));       // 64 KB
    __bf16*   h1b = h0b + (size_t)B_ * H_;                          // 64 KB
    float*    cb  = (float*)(h1b + (size_t)B_ * H_);                // 128 KB

    float* out  = (float*)d_out;
    float* outH = out + (size_t)B_ * T_ * H_;
    float* outC = outH + (size_t)B_ * H_;

    swizzle_B_kernel<<<512, 256, 0, stream>>>(Wi, Ui, Wf, Uf, Wo, Uo, Wc, Uc, Bsw);
    swizzle_X_kernel<<<4096, 256, 0, stream>>>(x, xf);
    init_state_kernel<<<128, 256, 0, stream>>>(h0, c0, h0b, cb, cnt);
    lstm_persistent_kernel<<<32, 128, 131072, stream>>>(xf, Bsw, bi, bf, bo, bc,
                                                        h0b, h1b, cb, out, outH, outC, cnt);
}
